// BiWMSA_67611375174269
// MI455X (gfx1250) — 8-bit matrix operand order repaired and checked by reading, compiled, not run on hardware
//
#include <hip/hip_runtime.h>

typedef int v8i __attribute__((ext_vector_type(8)));

#define WIN 8
#define HEADDIM 30
#define NHEADS 6
#define CIN 180
#define KPAD 192
#define NKV 360
#define NKV_TILES 23
#define IMG 192
#define GW 24
#define NWIN 576
#define NBATCH 2
#define SIM_SCALE 0.18257418583505536f
#define LOG_QUARTER -1.3862943611198906f
#define NEG_BIG -3.0e38f

#define WS_WKV 0
#define WS_WQ 70656
#define WS_WO 107520
#define WS_BTAB 144384

#define L_XS 0
#define L_YS 12288
#define L_QA 24576
#define L_KB 49152
#define L_VB 73728
#define L_OS 86016
#define L_S  0
#define L_P  16384
#define L_PM 20480
#define LDS_BYTES 98304

__device__ __forceinline__ signed char sgn8(float v) {
  return (signed char)((v > 0.f) - (v < 0.f));
}
__device__ __forceinline__ signed char sgn8i(int v) {
  return (signed char)((v > 0) - (v < 0));
}

__device__ __forceinline__ v8i load_A8(const signed char* base, int rowstride,
                                       int row0, int col0, int lane) {
  const int hl = lane >> 4, m = lane & 15;
  const signed char* rp = base + (row0 + m) * rowstride + col0;
  v8i a;
#pragma unroll
  for (int i = 0; i < 8; ++i) {
    int k = ((i >> 2) * 32) + (((i >> 1) & 1) * 16) + hl * 8 + (i & 1) * 4;
    a[i] = *reinterpret_cast<const int*>(rp + k);
  }
  return a;
}

__device__ __forceinline__ v8i load_B8(const signed char* base, int rowstride,
                                       int n0, int k0, int lane) {
  const int hl = lane >> 4, n = lane & 15;
  const signed char* rp = base + (n0 + n) * rowstride + k0;
  v8i b;
#pragma unroll
  for (int i = 0; i < 8; ++i) {
    int k = ((i >> 2) * 32) + (((i >> 1) & 1) * 16) + hl * 8 + (i & 1) * 4;
    b[i] = *reinterpret_cast<const int*>(rp + k);
  }
  return b;
}

__device__ __forceinline__ v8i wmma_ss(v8i a, v8i b, v8i c) {
  return __builtin_amdgcn_wmma_i32_16x16x64_iu8(true, a, true, b, c, false, false);
}
__device__ __forceinline__ v8i wmma_us(v8i a, v8i b, v8i c) {
  return __builtin_amdgcn_wmma_i32_16x16x64_iu8(false, a, true, b, c, false, false);
}

__device__ __forceinline__ unsigned int pack_sgn4(float4 v) {
  unsigned int r = (unsigned char)sgn8(v.x);
  r |= ((unsigned int)(unsigned char)sgn8(v.y)) << 8;
  r |= ((unsigned int)(unsigned char)sgn8(v.z)) << 16;
  r |= ((unsigned int)(unsigned char)sgn8(v.w)) << 24;
  return r;
}

__global__ void prep_weights(const float* __restrict__ w_kv,
                             const float* __restrict__ w_q,
                             const float* __restrict__ w_o,
                             signed char* __restrict__ ws8) {
  __shared__ float red[64];
  int r = blockIdx.x;
  const int tid = threadIdx.x;
  const float* src;
  signed char* dst;
  int nrows;
  if (r < 368)      { src = w_kv; dst = ws8 + WS_WKV; nrows = 360; }
  else if (r < 560) { src = w_q;  dst = ws8 + WS_WQ;  nrows = 180; r -= 368; }
  else              { src = w_o;  dst = ws8 + WS_WO;  nrows = 180; r -= 560; }
  signed char* drow = dst + r * KPAD;
  if (r >= nrows) {
    for (int c = tid; c < KPAD; c += 64) drow[c] = 0;
    return;
  }
  const float* srow = src + r * CIN;
  float s = 0.f;
  for (int c = tid; c < CIN; c += 64) s += srow[c];
  red[tid] = s;
  __syncthreads();
  for (int off = 32; off > 0; off >>= 1) {
    if (tid < off) red[tid] += red[tid + off];
    __syncthreads();
  }
  const float mean = red[0] * (1.f / (float)CIN);
  for (int c = tid; c < KPAD; c += 64)
    drow[c] = (c < CIN) ? sgn8(srow[c] - mean) : (signed char)0;
}

__global__ void prep_bias(const float* __restrict__ rel_pos, float* __restrict__ btab) {
  int idx = blockIdx.x * blockDim.x + threadIdx.x;
  int h = idx >> 12;
  int p = (idx >> 6) & 63;
  int q = idx & 63;
  int pi = p >> 3, pj = p & 7, qi = q >> 3, qj = q & 7;
  int r0 = pi - qi + 7, r1 = pj - qj + 7;
  btab[idx] = rel_pos[(h * 15 + r0) * 15 + r1];
}

__global__ __launch_bounds__(256) void biwmsa_main(
    const float* __restrict__ x, const float* __restrict__ y,
    const float* __restrict__ b_kv, const float* __restrict__ s_kv,
    const float* __restrict__ b_q, const float* __restrict__ s_q,
    const float* __restrict__ b_o, const float* __restrict__ s_o,
    const signed char* __restrict__ ws8, const float* __restrict__ btab,
    float* __restrict__ out) {
  extern __shared__ char smem[];
  signed char* Xs = (signed char*)(smem + L_XS);
  signed char* Ys = (signed char*)(smem + L_YS);
  signed char* QA = (signed char*)(smem + L_QA);
  signed char* KB = (signed char*)(smem + L_KB);
  signed char* VB = (signed char*)(smem + L_VB);
  signed char* OS = (signed char*)(smem + L_OS);
  float* S = (float*)(smem + L_S);
  unsigned char* P = (unsigned char*)(smem + L_P);
  float* PM = (float*)(smem + L_PM);

  const signed char* WKV = ws8 + WS_WKV;
  const signed char* WQ = ws8 + WS_WQ;
  const signed char* WO = ws8 + WS_WO;

  const int tid = threadIdx.x;
  const int lane = tid & 31;
  const int wave = tid >> 5;
  const int hl = lane >> 4;
  const int ln16 = lane & 15;

  const int blk = blockIdx.x;
  const int bb = blk / NWIN;
  const int w = blk % NWIN;
  const int wy = w / GW, wx = w % GW;

  {
    int* z = (int*)(smem + L_QA);
    const int nz = (L_OS + 64 * KPAD - L_QA) / 4;
    for (int i = tid; i < nz; i += 256) z[i] = 0;
  }

  for (int i = tid; i < 64 * (CIN / 4); i += 256) {
    int t = i / (CIN / 4), c4 = i - t * (CIN / 4);
    int ch = c4 * 4;
    int pi = t >> 3, pj = t & 7;
    int gr = wy * WIN + pi + (WIN / 2); if (gr >= IMG) gr -= IMG;
    int gc = wx * WIN + pj + (WIN / 2); if (gc >= IMG) gc -= IMG;
    size_t gidx = (((size_t)bb * IMG + gr) * IMG + gc) * CIN + ch;
    float4 vx = *reinterpret_cast<const float4*>(x + gidx);
    float4 vy = *reinterpret_cast<const float4*>(y + gidx);
    *reinterpret_cast<unsigned int*>(Xs + t * KPAD + ch) = pack_sgn4(vx);
    *reinterpret_cast<unsigned int*>(Ys + t * KPAD + ch) = pack_sgn4(vy);
  }
  for (int i = tid; i < 64 * (KPAD - CIN) / 4; i += 256) {
    int t = i / 3, c = i - t * 3;
    *reinterpret_cast<unsigned int*>(Xs + t * KPAD + CIN + c * 4) = 0u;
    *reinterpret_cast<unsigned int*>(Ys + t * KPAD + CIN + c * 4) = 0u;
  }
  __syncthreads();

  for (int nt = wave; nt < 12; nt += 8) {
    v8i bf0 = load_B8(WQ, KPAD, nt * 16, 0, lane);
    v8i bf1 = load_B8(WQ, KPAD, nt * 16, 64, lane);
    v8i bf2 = load_B8(WQ, KPAD, nt * 16, 128, lane);
    int n = nt * 16 + ln16;
    bool valid = (n < CIN);
    float sc = valid ? s_q[n] : 0.f;
    float bi = valid ? b_q[n] : 0.f;
    int h = valid ? (n / HEADDIM) : 0;
    int cc = n - h * HEADDIM;
#pragma unroll
    for (int mt = 0; mt < 4; ++mt) {
      v8i acc = {0, 0, 0, 0, 0, 0, 0, 0};
      acc = wmma_ss(load_A8(Ys, KPAD, mt * 16, 0, lane), bf0, acc);
      acc = wmma_ss(load_A8(Ys, KPAD, mt * 16, 64, lane), bf1, acc);
      acc = wmma_ss(load_A8(Ys, KPAD, mt * 16, 128, lane), bf2, acc);
      if (valid) {
#pragma unroll
        for (int r = 0; r < 8; ++r) {
          int token = mt * 16 + r + hl * 8;
          QA[(h * 64 + token) * 64 + cc] = sgn8(sc * (float)acc[r] + bi);
        }
      }
    }
  }

  for (int nt = wave; nt < NKV_TILES; nt += 8) {
    v8i bf0 = load_B8(WKV, KPAD, nt * 16, 0, lane);
    v8i bf1 = load_B8(WKV, KPAD, nt * 16, 64, lane);
    v8i bf2 = load_B8(WKV, KPAD, nt * 16, 128, lane);
    int n = nt * 16 + ln16;
    bool valid = (n < NKV);
    float sc = valid ? s_kv[n] : 0.f;
    float bi = valid ? b_kv[n] : 0.f;
#pragma unroll
    for (int mt = 0; mt < 4; ++mt) {
      v8i acc = {0, 0, 0, 0, 0, 0, 0, 0};
      acc = wmma_ss(load_A8(Xs, KPAD, mt * 16, 0, lane), bf0, acc);
      acc = wmma_ss(load_A8(Xs, KPAD, mt * 16, 64, lane), bf1, acc);
      acc = wmma_ss(load_A8(Xs, KPAD, mt * 16, 128, lane), bf2, acc);
      if (valid) {
#pragma unroll
        for (int r = 0; r < 8; ++r) {
          int token = mt * 16 + r + hl * 8;
          signed char s8 = sgn8(sc * (float)acc[r] + bi);
          if (n < CIN) {
            int h = n / HEADDIM, cc = n - h * HEADDIM;
            KB[(h * 64 + token) * 64 + cc] = s8;
          } else {
            int j = n - CIN;
            int h = j / HEADDIM, cc = j - h * HEADDIM;
            VB[(h * 32 + cc) * 64 + token] = s8;
          }
        }
      }
    }
  }
  __syncthreads();

  const bool lastCol = (wx == GW - 1);
  const bool lastRow = (wy == GW - 1);

  for (int h = 0; h < NHEADS; ++h) {
    const signed char* QAh = QA + h * 64 * 64;
    const signed char* KBh = KB + h * 64 * 64;
    const signed char* VBh = VB + h * 32 * 64;
    {
      int nt = wave & 3;
      v8i bf = load_B8(KBh, 64, nt * 16, 0, lane);
      int q = nt * 16 + ln16;
      int qi = q >> 3, qj = q & 7;
#pragma unroll
      for (int s2 = 0; s2 < 2; ++s2) {
        int mt = (wave >> 2) * 2 + s2;
        v8i a = load_A8(QAh, 64, mt * 16, 0, lane);
        v8i zero = {0, 0, 0, 0, 0, 0, 0, 0};
        v8i acc = wmma_ss(a, bf, zero);
#pragma unroll
        for (int r = 0; r < 8; ++r) {
          int p = mt * 16 + r + hl * 8;
          int pi = p >> 3, pj = p & 7;
          float f = (float)acc[r] * SIM_SCALE + btab[(h * 64 + p) * 64 + q];
          bool msk = (lastCol && ((pj < 4) != (qj < 4))) ||
                     (lastRow && ((pi < 4) != (qi < 4)));
          S[p * 64 + q] = msk ? NEG_BIG : f;
        }
      }
    }
    __syncthreads();
    {
      int p = tid >> 2, qq = tid & 3;
      const float* row = S + p * 64 + qq * 16;
      float mx = row[0];
#pragma unroll
      for (int j = 1; j < 16; ++j) mx = fmaxf(mx, row[j]);
      PM[p * 4 + qq] = mx;
    }
    __syncthreads();
    {
      int p = tid >> 2, qq = tid & 3;
      float m4 = fmaxf(fmaxf(PM[p * 4 + 0], PM[p * 4 + 1]),
                       fmaxf(PM[p * 4 + 2], PM[p * 4 + 3]));
      float th = m4 + LOG_QUARTER;
      const float* row = S + p * 64 + qq * 16;
      unsigned char* pr = P + p * 64 + qq * 16;
#pragma unroll
      for (int j = 0; j < 16; ++j) pr[j] = (row[j] >= th) ? 1 : 0;
    }
    __syncthreads();
    {
      int mt = wave >> 1, nt = wave & 1;
      v8i a = load_A8((const signed char*)P, 64, mt * 16, 0, lane);
      v8i b = load_B8(VBh, 64, nt * 16, 0, lane);
      v8i zero = {0, 0, 0, 0, 0, 0, 0, 0};
      v8i acc = wmma_us(a, b, zero);
      int cc = nt * 16 + ln16;
      if (cc < HEADDIM) {
#pragma unroll
        for (int r = 0; r < 8; ++r) {
          int token = mt * 16 + r + hl * 8;
          OS[token * KPAD + h * HEADDIM + cc] = sgn8i(acc[r]);
        }
      }
    }
  }
  __syncthreads();

  for (int nt = wave; nt < 12; nt += 8) {
    v8i bf0 = load_B8(WO, KPAD, nt * 16, 0, lane);
    v8i bf1 = load_B8(WO, KPAD, nt * 16, 64, lane);
    v8i bf2 = load_B8(WO, KPAD, nt * 16, 128, lane);
    int n = nt * 16 + ln16;
    bool valid = (n < CIN);
    float sc = valid ? s_o[n] : 0.f;
    float bi = valid ? b_o[n] : 0.f;
#pragma unroll
    for (int mt = 0; mt < 4; ++mt) {
      v8i acc = {0, 0, 0, 0, 0, 0, 0, 0};
      acc = wmma_ss(load_A8(OS, KPAD, mt * 16, 0, lane), bf0, acc);
      acc = wmma_ss(load_A8(OS, KPAD, mt * 16, 64, lane), bf1, acc);
      acc = wmma_ss(load_A8(OS, KPAD, mt * 16, 128, lane), bf2, acc);
      if (valid) {
#pragma unroll
        for (int r = 0; r < 8; ++r) {
          int token = mt * 16 + r + hl * 8;
          int pi = token >> 3, pj = token & 7;
          int gr = wy * WIN + pi + (WIN / 2); if (gr >= IMG) gr -= IMG;
          int gc = wx * WIN + pj + (WIN / 2); if (gc >= IMG) gc -= IMG;
          size_t gidx = (((size_t)bb * IMG + gr) * IMG + gc) * CIN + n;
          out[gidx] = sc * (float)acc[r] + bi;
        }
      }
    }
  }
}

extern "C" void kernel_launch(void* const* d_in, const int* in_sizes, int n_in,
                              void* d_out, int out_size, void* d_ws, size_t ws_size,
                              hipStream_t stream) {
  (void)in_sizes; (void)n_in; (void)out_size; (void)ws_size;
  const float* x = (const float*)d_in[0];
  const float* y = (const float*)d_in[1];
  const float* w_kv = (const float*)d_in[2];
  const float* b_kv = (const float*)d_in[3];
  const float* s_kv = (const float*)d_in[4];
  const float* w_q = (const float*)d_in[5];
  const float* b_q = (const float*)d_in[6];
  const float* s_q = (const float*)d_in[7];
  const float* w_o = (const float*)d_in[8];
  const float* b_o = (const float*)d_in[9];
  const float* s_o = (const float*)d_in[10];
  const float* rel_pos = (const float*)d_in[11];

  signed char* ws8 = (signed char*)d_ws;
  float* btab = (float*)((char*)d_ws + WS_BTAB);

  prep_weights<<<752, 64, 0, stream>>>(w_kv, w_q, w_o, ws8);
  prep_bias<<<96, 256, 0, stream>>>(rel_pos, btab);
  biwmsa_main<<<NBATCH * NWIN, 256, LDS_BYTES, stream>>>(
      x, y, b_kv, s_kv, b_q, s_q, b_o, s_o, ws8, btab, (float*)d_out);
}
